// RegularDecoder_66520453481161
// MI455X (gfx1250) — matrix operand order repaired and checked by reading, not compiled, not run on hardware
//
#include <hip/hip_runtime.h>
#include <hip/hip_bf16.h>

#define HID   1024
#define OUTD  512
#define BATCH 512
#define SEQ   128

typedef __attribute__((ext_vector_type(16))) __bf16 v16bf;
typedef __attribute__((ext_vector_type(8)))  __bf16 v8bf;
typedef __attribute__((ext_vector_type(8)))  float  v8f;

union V16U { v16bf v; v8bf h[2]; };

__device__ __forceinline__ float sigmoidf_(float x) {
    return 1.0f / (1.0f + __expf(-x));
}

// ---------------------------------------------------------------------------
// f32 -> bf16 elementwise convert
// ---------------------------------------------------------------------------
__global__ void cvt_f32_bf16(const float* __restrict__ src, __bf16* __restrict__ dst, int n) {
    int i = blockIdx.x * blockDim.x + threadIdx.x;
    if (i < n) dst[i] = (__bf16)src[i];
}

// combined bias for the input projection: b_ih + b_hh
__global__ void bias_comb(const float* __restrict__ a, const float* __restrict__ b,
                          float* __restrict__ out, int n) {
    int i = blockIdx.x * blockDim.x + threadIdx.x;
    if (i < n) out[i] = a[i] + b[i];
}

// ---------------------------------------------------------------------------
// Generic bf16 WMMA GEMM: D[M,N] = A[M,K=1024] @ B[N,K=1024]^T + bias[N]
// A, B row-major, K-contiguous. Block = 4 waves (128 thr); block tile 32x256.
// Each wave owns a 32x64 slab: acc[2 M-subtiles][4 N-subtiles], K-step 32.
// Per k-chunk all 12 b128 loads (2 A + 4 B tiles) are issued before the 8
// WMMAs so the compiler can use partial s_wait_loadcnt and overlap latency.
// ---------------------------------------------------------------------------
__global__ __launch_bounds__(128)
void gemm_bf16(const __bf16* __restrict__ A, const __bf16* __restrict__ B,
               const float* __restrict__ bias, float* __restrict__ D, int N) {
    const int lane = threadIdx.x & 31;
    const int wave = threadIdx.x >> 5;
    const int m0   = blockIdx.x * 32;
    const int n0   = blockIdx.y * 256 + wave * 64;
    const int m    = lane & 15;
    const int hi   = lane >> 4;          // 0: lanes 0-15, 1: lanes 16-31

    v8f acc[2][4] = {};

    // A: lane holds row m0+mt*16+m, 16B chunks at K = kA and kA+16 (kA = hi?8:0)
    const __bf16* arow[2];
#pragma unroll
    for (int mt = 0; mt < 2; ++mt)
        arow[mt] = A + (size_t)(m0 + mt * 16 + m) * HID + (hi ? 8 : 0);
    // B: lane holds W row (n0+s*16+m), 16B chunks at K = kB and kB+16 (kB = hi?8:0)
    const __bf16* brow[4];
#pragma unroll
    for (int s = 0; s < 4; ++s)
        brow[s] = B + (size_t)(n0 + s * 16 + m) * HID + (hi ? 8 : 0);

    for (int k0 = 0; k0 < HID; k0 += 32) {
        V16U a[2], b[4];
#pragma unroll
        for (int mt = 0; mt < 2; ++mt) {
            a[mt].h[0] = *(const v8bf*)(arow[mt] + k0);
            a[mt].h[1] = *(const v8bf*)(arow[mt] + k0 + 16);
        }
#pragma unroll
        for (int s = 0; s < 4; ++s) {
            b[s].h[0] = *(const v8bf*)(brow[s] + k0);
            b[s].h[1] = *(const v8bf*)(brow[s] + k0 + 16);
        }
#pragma unroll
        for (int s = 0; s < 4; ++s)
#pragma unroll
            for (int mt = 0; mt < 2; ++mt)
                acc[mt][s] = __builtin_amdgcn_wmma_f32_16x16x32_bf16(
                    false, a[mt].v, false, b[s].v, (short)0, acc[mt][s], false, false);
    }

    // D layout: lane -> col n0+s*16+m ; vgpr r -> row m0 + mt*16 + r + hi*8
#pragma unroll
    for (int s = 0; s < 4; ++s) {
        int col = n0 + s * 16 + m;
        float bv = bias[col];
#pragma unroll
        for (int mt = 0; mt < 2; ++mt) {
#pragma unroll
            for (int r = 0; r < 8; ++r) {
                int row = m0 + mt * 16 + r + (hi ? 8 : 0);
                D[(size_t)row * N + col] = acc[mt][s][r] + bv;
            }
        }
    }
}

// ---------------------------------------------------------------------------
// Fused LSTM step. Block region: 64 batch rows x 64 hidden cols across all 4
// gate sections (i,f,g,o). 8 waves: wave = gate + 4*msub; each wave computes a
// 32x64 slab of its gate via WMMA, gate tiles meet in LDS, then the pointwise
// cell. Grid: (BATCH/64, HID/64) = (8, 16). Block: 256 threads.
// ---------------------------------------------------------------------------
__global__ __launch_bounds__(256)
void lstm_step(const __bf16* __restrict__ hprev, const __bf16* __restrict__ Whh,
               const float* __restrict__ xproj, float* __restrict__ cstate,
               __bf16* __restrict__ hout, int first) {
    __shared__ float gsm[4][64][64];          // 64 KB: i,f,g,o tiles

    const int lane = threadIdx.x & 31;
    const int wave = threadIdx.x >> 5;
    const int gate = wave & 3;                // gate section (i,f,g,o)
    const int msub = wave >> 2;               // 0 or 1: which 32-row half
    const int b0   = blockIdx.x * 64;         // batch tile
    const int j0   = blockIdx.y * 64;         // hidden-col tile within a gate
    const int m    = lane & 15;
    const int hi   = lane >> 4;
    const int mw   = b0 + msub * 32;          // wave's first batch row

    v8f acc[2][4] = {};

    if (!first) {
        const __bf16* arow[2];
#pragma unroll
        for (int mt = 0; mt < 2; ++mt)
            arow[mt] = hprev + (size_t)(mw + mt * 16 + m) * HID + (hi ? 8 : 0);
        const __bf16* brow[4];
#pragma unroll
        for (int s = 0; s < 4; ++s)
            brow[s] = Whh + (size_t)(gate * HID + j0 + s * 16 + m) * HID + (hi ? 8 : 0);

        for (int k0 = 0; k0 < HID; k0 += 32) {
            V16U a[2], b[4];
#pragma unroll
            for (int mt = 0; mt < 2; ++mt) {
                a[mt].h[0] = *(const v8bf*)(arow[mt] + k0);
                a[mt].h[1] = *(const v8bf*)(arow[mt] + k0 + 16);
            }
#pragma unroll
            for (int s = 0; s < 4; ++s) {
                b[s].h[0] = *(const v8bf*)(brow[s] + k0);
                b[s].h[1] = *(const v8bf*)(brow[s] + k0 + 16);
            }
#pragma unroll
            for (int s = 0; s < 4; ++s)
#pragma unroll
                for (int mt = 0; mt < 2; ++mt)
                    acc[mt][s] = __builtin_amdgcn_wmma_f32_16x16x32_bf16(
                        false, a[mt].v, false, b[s].v, (short)0, acc[mt][s], false, false);
        }
    }

    // add x_proj, park gate tiles in LDS
#pragma unroll
    for (int s = 0; s < 4; ++s) {
        int nl = s * 16 + m;
#pragma unroll
        for (int mt = 0; mt < 2; ++mt) {
#pragma unroll
            for (int r = 0; r < 8; ++r) {
                int ml = msub * 32 + mt * 16 + r + (hi ? 8 : 0);
                gsm[gate][ml][nl] =
                    acc[mt][s][r] +
                    xproj[(size_t)(b0 + ml) * (4 * HID) + gate * HID + j0 + nl];
            }
        }
    }
    __syncthreads();

    // pointwise LSTM cell over the 64x64 region (16 elems/thread)
    for (int e = threadIdx.x; e < 64 * 64; e += 256) {
        int mm = e >> 6, nl = e & 63;
        float iv = sigmoidf_(gsm[0][mm][nl]);
        float fv = sigmoidf_(gsm[1][mm][nl]);
        float gv = tanhf(gsm[2][mm][nl]);
        float ov = sigmoidf_(gsm[3][mm][nl]);
        size_t idx = (size_t)(b0 + mm) * HID + j0 + nl;
        float cp = first ? 0.0f : cstate[idx];
        float cn = fv * cp + iv * gv;
        cstate[idx] = cn;
        hout[idx] = (__bf16)(ov * tanhf(cn));
    }
}

// ---------------------------------------------------------------------------
extern "C" void kernel_launch(void* const* d_in, const int* in_sizes, int n_in,
                              void* d_out, int out_size, void* d_ws, size_t ws_size,
                              hipStream_t stream) {
    const float* C_in  = (const float*)d_in[0];
    // d_in[1]=t, d_in[2]=mask : unused by the reference forward
    const float* W_ih  = (const float*)d_in[3];
    const float* W_hh  = (const float*)d_in[4];
    const float* b_ih  = (const float*)d_in[5];
    const float* b_hh  = (const float*)d_in[6];
    const float* W_lin = (const float*)d_in[7];
    const float* b_lin = (const float*)d_in[8];
    float* out = (float*)d_out;

    // ---- workspace carve ----
    char* ws = (char*)d_ws;
    __bf16* Wih16  = (__bf16*)ws; ws += (size_t)4 * HID * HID * 2;    // 8 MB
    __bf16* Whh16  = (__bf16*)ws; ws += (size_t)4 * HID * HID * 2;    // 8 MB
    __bf16* Wlin16 = (__bf16*)ws; ws += (size_t)OUTD * HID * 2;       // 1 MB
    __bf16* C16    = (__bf16*)ws; ws += (size_t)BATCH * HID * 2;      // 1 MB
    float*  biasc  = (float*)ws;  ws += (size_t)4 * HID * 4;          // 16 KB
    float*  xproj  = (float*)ws;  ws += (size_t)BATCH * 4 * HID * 4;  // 8 MB
    float*  cst    = (float*)ws;  ws += (size_t)BATCH * HID * 4;      // 2 MB
    __bf16* hring  = (__bf16*)ws; ws += (size_t)8 * BATCH * HID * 2;  // 8 MB ring of h_t

    // ---- one-time conversions to bf16 ----
    {
        int n;
        n = 4 * HID * HID;
        cvt_f32_bf16<<<(n + 255) / 256, 256, 0, stream>>>(W_ih, Wih16, n);
        cvt_f32_bf16<<<(n + 255) / 256, 256, 0, stream>>>(W_hh, Whh16, n);
        n = OUTD * HID;
        cvt_f32_bf16<<<(n + 255) / 256, 256, 0, stream>>>(W_lin, Wlin16, n);
        n = BATCH * HID;
        cvt_f32_bf16<<<(n + 255) / 256, 256, 0, stream>>>(C_in, C16, n);
        bias_comb<<<(4 * HID + 255) / 256, 256, 0, stream>>>(b_ih, b_hh, biasc, 4 * HID);
    }

    // ---- input projection: xproj[512,4096] = C @ W_ih^T + (b_ih + b_hh) ----
    {
        dim3 grid(BATCH / 32, (4 * HID) / 256);
        gemm_bf16<<<grid, 128, 0, stream>>>(C16, Wih16, biasc, xproj, 4 * HID);
    }

    // ---- 128 sequential steps; batched output projection every 8 steps ----
    const size_t hslot = (size_t)BATCH * HID;
    dim3 gstep(BATCH / 64, HID / 64);
    dim3 gproj((8 * BATCH) / 32, OUTD / 256);
    for (int t = 0; t < SEQ; ++t) {
        const __bf16* hin  = hring + ((t - 1) & 7) * hslot;   // unused when t==0
        __bf16*       hnew = hring + (t & 7) * hslot;
        lstm_step<<<gstep, 256, 0, stream>>>(hin, Whh16, xproj, cst, hnew, t == 0);
        if ((t & 7) == 7) {
            // ring slots 0..7 hold h_{t-7..t} in order; project all 8 steps at once
            gemm_bf16<<<gproj, 128, 0, stream>>>(
                hring, Wlin16, b_lin,
                out + (size_t)(t - 7) * BATCH * OUTD, OUTD);
        }
    }
}
